// Bottleneck_18588618457246
// MI455X (gfx1250) — matrix operand order repaired and checked by reading, not compiled, not run on hardware
//
#include <hip/hip_runtime.h>
#include <hip/hip_bf16.h>
#include <math.h>

#define B_   4
#define C_   512
#define E_   4096
#define T_   2048
#define KSP  512
#define ET   (E_*T_)            // 2^23
#define WGPB 512                // (E_/128)*(T_/128) GEMM workgroups per batch
#define CAND_CAP 8192
#define LDT  40                 // padded LDS row stride (halfs): 80B, 16B aligned
#define TILEH (128*LDT)         // halfs per LDS tile buffer
#define TILEB (TILEH*2)         // bytes per LDS tile buffer (10240)

typedef _Float16 v8h  __attribute__((ext_vector_type(8)));
typedef _Float16 v16h __attribute__((ext_vector_type(16)));
typedef float    v8f  __attribute__((ext_vector_type(8)));

#define CONCAT16(a,b) __builtin_shufflevector(a,b,0,1,2,3,4,5,6,7,8,9,10,11,12,13,14,15)

__device__ inline unsigned ordkey(float f) {
    unsigned u = __float_as_uint(f);
    return (u & 0x80000000u) ? ~u : (u | 0x80000000u);   // monotone float->uint
}

// Async global->LDS copy of one 16B chunk (CDNA5 GLOBAL_LOAD_ASYNC_TO_LDS_B128,
// ASYNCcnt-tracked). LDS dest = LDS byte offset in a VGPR.
__device__ inline void async_cp16(unsigned ldsAddr, unsigned long long gAddr) {
    asm volatile("global_load_async_to_lds_b128 %0, %1, off"
                 :: "v"(ldsAddr), "v"(gAddr) : "memory");
}
__device__ inline void async_wait0() {
    asm volatile("s_wait_asynccnt 0" ::: "memory");
}

// ---------------------------------------------------------------------------
// Precompute: split f32 -> (hi, lo) f16 pairs. hi+lo carries ~22 mantissa bits.
// ---------------------------------------------------------------------------
__global__ void split_w(const float* __restrict__ W,
                        _Float16* __restrict__ WH, _Float16* __restrict__ WL, int n)
{
    for (int i = blockIdx.x*blockDim.x + threadIdx.x; i < n; i += gridDim.x*blockDim.x) {
        const float f = W[i];
        const _Float16 h = (_Float16)f;
        WH[i] = h; WL[i] = (_Float16)(f - (float)h);
    }
}

// x[B][C][T] -> xT[B][T][C] split into hi/lo f16 (LDS-tiled transpose)
__global__ __launch_bounds__(256) void split_transpose_x(
    const float* __restrict__ x, _Float16* __restrict__ xTH, _Float16* __restrict__ xTL)
{
    __shared__ float tile[32][33];
    const int b  = blockIdx.z;
    const int c0 = blockIdx.y * 32;
    const int t0 = blockIdx.x * 32;
    const int tx = threadIdx.x & 31;
    const int ty = threadIdx.x >> 5;          // 0..7
    #pragma unroll
    for (int k = 0; k < 4; ++k) {
        const int c = ty + 8*k;
        tile[c][tx] = x[((size_t)b*C_ + c0 + c)*T_ + t0 + tx];
    }
    __syncthreads();
    #pragma unroll
    for (int k = 0; k < 4; ++k) {
        const int t = ty + 8*k;
        const float f = tile[tx][t];
        const _Float16 h = (_Float16)f;
        const size_t o = ((size_t)b*T_ + t0 + t)*C_ + c0 + tx;
        xTH[o] = h; xTL[o] = (_Float16)(f - (float)h);
    }
}

// ---------------------------------------------------------------------------
// GEMM epilogue (shared by both GEMM kernels): bias, write mask, WG stats
// ---------------------------------------------------------------------------
__device__ inline void gemm_epilogue(
    v8f acc[4][2], const float* __restrict__ bm, float* __restrict__ mask,
    float* __restrict__ wgMax, float* __restrict__ wgSum, float* red,
    int tid, int wm, int wn, int half, int l16, int e0, int t0, int b, int gdx)
{
    float bv[4][8];
    #pragma unroll
    for (int fm = 0; fm < 4; ++fm)
        #pragma unroll
        for (int r = 0; r < 8; ++r)
            bv[fm][r] = bm[e0 + wm*64 + fm*16 + 8*half + r];

    float mx = -3.4e38f;
    #pragma unroll
    for (int fm = 0; fm < 4; ++fm)
        #pragma unroll
        for (int fn = 0; fn < 2; ++fn)
            #pragma unroll
            for (int r = 0; r < 8; ++r)
                mx = fmaxf(mx, acc[fm][fn][r] + bv[fm][r]);

    red[tid] = mx; __syncthreads();
    for (int s = 128; s > 0; s >>= 1) { if (tid < s) red[tid] = fmaxf(red[tid], red[tid+s]); __syncthreads(); }
    const float wmax = red[0];
    __syncthreads();

    float se = 0.f;
    #pragma unroll
    for (int fm = 0; fm < 4; ++fm)
        #pragma unroll
        for (int r = 0; r < 8; ++r) {
            const int e = e0 + wm*64 + fm*16 + 8*half + r;    // C/D layout: M = r + 8*half
            #pragma unroll
            for (int fn = 0; fn < 2; ++fn) {
                const int t = t0 + wn*32 + fn*16 + l16;       // N = lane&15
                const float v = acc[fm][fn][r] + bv[fm][r];
                se += __expf(v - wmax);
                mask[((size_t)b*E_ + e)*T_ + t] = v;
            }
        }
    red[tid] = se; __syncthreads();
    for (int s = 128; s > 0; s >>= 1) { if (tid < s) red[tid] += red[tid+s]; __syncthreads(); }
    if (tid == 0) {
        const int wg = blockIdx.y * gdx + blockIdx.x;
        wgMax[b*WGPB + wg] = wmax;
        wgSum[b*WGPB + wg] = red[0];
    }
}

// ---------------------------------------------------------------------------
// Primary GEMM: pre-split f16 operands, async global->LDS staging, double
// buffered. 24 v_wmma per k-step (split precision: AhBh + AhBl + AlBh).
// Unified LDS array smem[tile][buf][...]: tile 0=Ah 1=Al 2=Bh 3=Bl.
// ---------------------------------------------------------------------------
__global__ __launch_bounds__(256) void mask_gemm_async(
    const _Float16* __restrict__ WmH, const _Float16* __restrict__ WmL,
    const _Float16* __restrict__ xTH, const _Float16* __restrict__ xTL,
    const float* __restrict__ bm, float* __restrict__ mask,
    float* __restrict__ wgMax, float* __restrict__ wgSum)
{
    __shared__ _Float16 smem[4][2][TILEH];    // 80 KB
    __shared__ float red[256];

    const int tid  = threadIdx.x;
    const int lane = tid & 31;
    const int w    = tid >> 5;
    const int wm   = w & 1;
    const int wn   = w >> 1;
    const int half = lane >> 4;
    const int l16  = lane & 15;

    const int t0 = blockIdx.x * 128;
    const int e0 = blockIdx.y * 128;
    const int b  = blockIdx.z;

    // Per-thread static copy descriptors: 4 tiles * 512 chunks of 16B, 8/thread.
    // Chunk c: tile = c>>9, j = c&511, row = j>>2, cc = (j&3)*8 halfs.
    const unsigned ldsBase = (unsigned)(uintptr_t)(&smem[0][0][0]);
    const _Float16* gb[8];
    unsigned lo[8];
    {
        const _Float16* tbase[4] = { WmH + (size_t)e0 * C_, WmL + (size_t)e0 * C_,
                                     xTH + ((size_t)b*T_ + t0) * C_, xTL + ((size_t)b*T_ + t0) * C_ };
        #pragma unroll
        for (int q = 0; q < 8; ++q) {
            const int chunk = tid + (q << 8);
            const int tile = chunk >> 9;
            const int j    = chunk & 511;
            const int row  = j >> 2;
            const int cc   = (j & 3) << 3;                  // halfs
            gb[q] = tbase[tile] + (size_t)row * C_ + cc;
            lo[q] = ldsBase + 2u * (unsigned)((tile*2)*TILEH + row*LDT + cc);  // buf 0
        }
    }
    auto issue = [&](int c0, int buf) {
        #pragma unroll
        for (int q = 0; q < 8; ++q)
            async_cp16(lo[q] + (unsigned)buf * TILEB,
                       (unsigned long long)(uintptr_t)(gb[q] + c0));
    };

    v8f acc[4][2];
    #pragma unroll
    for (int fm = 0; fm < 4; ++fm)
        #pragma unroll
        for (int fn = 0; fn < 2; ++fn)
            #pragma unroll
            for (int r = 0; r < 8; ++r) acc[fm][fn][r] = 0.f;

    issue(0, 0);
    for (int ks = 0; ks < C_/32; ++ks) {
        const int buf = ks & 1;
        async_wait0();                 // my copies into `buf` have landed in LDS
        __syncthreads();               // everyone's have; prior compute on buf^1 drained
        if (ks + 1 < C_/32) issue((ks+1)*32, buf ^ 1);

        v16h aH[4], aL[4], bH[2], bL[2];
        #pragma unroll
        for (int fm = 0; fm < 4; ++fm) {
            const int row  = wm*64 + fm*16 + l16;
            const int koff = half * 8;
            const _Float16* pa = &smem[0][buf][row*LDT + koff];
            aH[fm] = CONCAT16(*(const v8h*)pa, *(const v8h*)(pa + 16));
            const _Float16* pl = &smem[1][buf][row*LDT + koff];
            aL[fm] = CONCAT16(*(const v8h*)pl, *(const v8h*)(pl + 16));
        }
        #pragma unroll
        for (int fn = 0; fn < 2; ++fn) {
            const int col = wn*32 + fn*16 + l16;
            const int kb  = half * 8;
            const _Float16* pb = &smem[2][buf][col*LDT + kb];
            bH[fn] = CONCAT16(*(const v8h*)pb, *(const v8h*)(pb + 16));
            const _Float16* pq = &smem[3][buf][col*LDT + kb];
            bL[fn] = CONCAT16(*(const v8h*)pq, *(const v8h*)(pq + 16));
        }
        #pragma unroll
        for (int fm = 0; fm < 4; ++fm)
            #pragma unroll
            for (int fn = 0; fn < 2; ++fn) {
                acc[fm][fn] = __builtin_amdgcn_wmma_f32_16x16x32_f16(
                    false, aH[fm], false, bH[fn], (short)0, acc[fm][fn], false, false);
                acc[fm][fn] = __builtin_amdgcn_wmma_f32_16x16x32_f16(
                    false, aH[fm], false, bL[fn], (short)0, acc[fm][fn], false, false);
                acc[fm][fn] = __builtin_amdgcn_wmma_f32_16x16x32_f16(
                    false, aL[fm], false, bH[fn], (short)0, acc[fm][fn], false, false);
            }
    }
    __syncthreads();
    gemm_epilogue(acc, bm, mask, wgMax, wgSum, red, tid, wm, wn, half, l16, e0, t0, b, gridDim.x);
}

// ---------------------------------------------------------------------------
// Fallback GEMM (used when ws can't hold pre-split operands): f32 inputs,
// in-loop split, single-buffered LDS staging.
// ---------------------------------------------------------------------------
__global__ __launch_bounds__(256) void mask_gemm(
    const float* __restrict__ x, const float* __restrict__ Wm,
    const float* __restrict__ bm, float* __restrict__ mask,
    float* __restrict__ wgMax, float* __restrict__ wgSum)
{
    __shared__ _Float16 Ah[128*LDT];
    __shared__ _Float16 Al[128*LDT];
    __shared__ _Float16 Bh[128*LDT];
    __shared__ _Float16 Bl[128*LDT];
    __shared__ float red[256];

    const int tid  = threadIdx.x;
    const int lane = tid & 31;
    const int w    = tid >> 5;
    const int wm   = w & 1;
    const int wn   = w >> 1;
    const int half = lane >> 4;
    const int l16  = lane & 15;

    const int t0 = blockIdx.x * 128;
    const int e0 = blockIdx.y * 128;
    const int b  = blockIdx.z;
    const float* xb = x + (size_t)b * C_ * T_;

    v8f acc[4][2];
    #pragma unroll
    for (int fm = 0; fm < 4; ++fm)
        #pragma unroll
        for (int fn = 0; fn < 2; ++fn)
            #pragma unroll
            for (int r = 0; r < 8; ++r) acc[fm][fn][r] = 0.f;

    for (int c0 = 0; c0 < C_; c0 += 32) {
        {
            const int r0 = tid >> 3;
            const int cc = (tid & 7) << 2;
            #pragma unroll
            for (int it = 0; it < 4; ++it) {
                const int row = it*32 + r0;
                const float4 v = *(const float4*)(Wm + (size_t)(e0+row)*C_ + c0 + cc);
                float f; _Float16 h;
                f=v.x; h=(_Float16)f; Ah[row*LDT+cc+0]=h; Al[row*LDT+cc+0]=(_Float16)(f-(float)h);
                f=v.y; h=(_Float16)f; Ah[row*LDT+cc+1]=h; Al[row*LDT+cc+1]=(_Float16)(f-(float)h);
                f=v.z; h=(_Float16)f; Ah[row*LDT+cc+2]=h; Al[row*LDT+cc+2]=(_Float16)(f-(float)h);
                f=v.w; h=(_Float16)f; Ah[row*LDT+cc+3]=h; Al[row*LDT+cc+3]=(_Float16)(f-(float)h);
            }
        }
        {
            const int cr = tid >> 5;
            const int tc = (tid & 31) << 2;
            #pragma unroll
            for (int it = 0; it < 4; ++it) {
                const int c = it*8 + cr;
                const float4 v = *(const float4*)(xb + (size_t)(c0+c)*T_ + t0 + tc);
                float f; _Float16 h;
                f=v.x; h=(_Float16)f; Bh[(tc+0)*LDT+c]=h; Bl[(tc+0)*LDT+c]=(_Float16)(f-(float)h);
                f=v.y; h=(_Float16)f; Bh[(tc+1)*LDT+c]=h; Bl[(tc+1)*LDT+c]=(_Float16)(f-(float)h);
                f=v.z; h=(_Float16)f; Bh[(tc+2)*LDT+c]=h; Bl[(tc+2)*LDT+c]=(_Float16)(f-(float)h);
                f=v.w; h=(_Float16)f; Bh[(tc+3)*LDT+c]=h; Bl[(tc+3)*LDT+c]=(_Float16)(f-(float)h);
            }
        }
        __syncthreads();

        v16h aH[4], aL[4], bH[2], bL[2];
        #pragma unroll
        for (int fm = 0; fm < 4; ++fm) {
            const int row  = wm*64 + fm*16 + l16;
            const int koff = half * 8;
            const _Float16* pa = &Ah[row*LDT + koff];
            aH[fm] = CONCAT16(*(const v8h*)pa, *(const v8h*)(pa + 16));
            const _Float16* pl = &Al[row*LDT + koff];
            aL[fm] = CONCAT16(*(const v8h*)pl, *(const v8h*)(pl + 16));
        }
        #pragma unroll
        for (int fn = 0; fn < 2; ++fn) {
            const int col = wn*32 + fn*16 + l16;
            const int kb  = half * 8;
            const _Float16* pb = &Bh[col*LDT + kb];
            bH[fn] = CONCAT16(*(const v8h*)pb, *(const v8h*)(pb + 16));
            const _Float16* pq = &Bl[col*LDT + kb];
            bL[fn] = CONCAT16(*(const v8h*)pq, *(const v8h*)(pq + 16));
        }
        #pragma unroll
        for (int fm = 0; fm < 4; ++fm)
            #pragma unroll
            for (int fn = 0; fn < 2; ++fn) {
                acc[fm][fn] = __builtin_amdgcn_wmma_f32_16x16x32_f16(
                    false, aH[fm], false, bH[fn], (short)0, acc[fm][fn], false, false);
                acc[fm][fn] = __builtin_amdgcn_wmma_f32_16x16x32_f16(
                    false, aH[fm], false, bL[fn], (short)0, acc[fm][fn], false, false);
                acc[fm][fn] = __builtin_amdgcn_wmma_f32_16x16x32_f16(
                    false, aL[fm], false, bH[fn], (short)0, acc[fm][fn], false, false);
            }
        __syncthreads();
    }
    gemm_epilogue(acc, bm, mask, wgMax, wgSum, red, tid, wm, wn, half, l16, e0, t0, b, gridDim.x);
}

// ---------------------------------------------------------------------------
// Kernel: combine per-WG (max,sum) -> per-batch global max and Z
// ---------------------------------------------------------------------------
__global__ __launch_bounds__(256) void softmax_stats(
    const float* __restrict__ wgMax, const float* __restrict__ wgSum,
    float* __restrict__ bMax, float* __restrict__ bZ)
{
    const int b = blockIdx.x, tid = threadIdx.x;
    __shared__ float red[256];
    float m = -3.4e38f;
    for (int i = tid; i < WGPB; i += 256) m = fmaxf(m, wgMax[b*WGPB + i]);
    red[tid] = m; __syncthreads();
    for (int s = 128; s > 0; s >>= 1) { if (tid < s) red[tid] = fmaxf(red[tid], red[tid+s]); __syncthreads(); }
    const float M = red[0];
    __syncthreads();
    float z = 0.f;
    for (int i = tid; i < WGPB; i += 256) z += wgSum[b*WGPB + i] * expf(wgMax[b*WGPB + i] - M);
    red[tid] = z; __syncthreads();
    for (int s = 128; s > 0; s >>= 1) { if (tid < s) red[tid] += red[tid+s]; __syncthreads(); }
    if (tid == 0) { bMax[b] = M; bZ[b] = red[0]; }
}

// ---------------------------------------------------------------------------
// Top-k machinery: 16-bit prefix histogram over order-preserving keys
// ---------------------------------------------------------------------------
__global__ void zero_u32(unsigned* __restrict__ p, int n) {
    for (int i = blockIdx.x*blockDim.x + threadIdx.x; i < n; i += gridDim.x*blockDim.x) p[i] = 0u;
}

__global__ void hist_kernel(const float* __restrict__ mask, unsigned* __restrict__ hist) {
    const long total = (long)B_ * ET;
    for (long i = (long)blockIdx.x*blockDim.x + threadIdx.x; i < total; i += (long)gridDim.x*blockDim.x) {
        const float v = mask[i];
        const int b = (int)(i >> 23);                       // ET = 2^23
        atomicAdd(&hist[(b << 16) + (ordkey(v) >> 16)], 1u);
    }
}

__global__ __launch_bounds__(256) void find_threshold(
    const unsigned* __restrict__ hist, unsigned* __restrict__ thrBin, unsigned* __restrict__ need)
{
    const int b = blockIdx.x, tid = threadIdx.x;
    __shared__ unsigned csum[256];
    const unsigned* h = hist + (b << 16);
    unsigned s = 0;
    for (int j = 0; j < 256; ++j) s += h[65535 - (tid*256 + j)];   // descending chunks
    csum[tid] = s; __syncthreads();
    if (tid == 0) {
        unsigned cum = 0; int c = 0;
        for (; c < 256; ++c) { if (cum + csum[c] >= KSP) break; cum += csum[c]; }
        unsigned kprev = cum, tb = 0;
        for (int j = 0; j < 256; ++j) {
            const int bin = 65535 - (c*256 + j);
            const unsigned cnt = h[bin];
            if (kprev + cnt >= KSP) { tb = (unsigned)bin; break; }
            kprev += cnt;
        }
        thrBin[b] = tb;
        need[b]   = KSP - kprev;
    }
}

__global__ void collect(const float* __restrict__ mask, const unsigned* __restrict__ thrBin,
                        unsigned* __restrict__ selCnt, unsigned* __restrict__ selIdx, float* __restrict__ selVal,
                        unsigned* __restrict__ candCnt, unsigned* __restrict__ candIdx, float* __restrict__ candVal)
{
    const long total = (long)B_ * ET;
    for (long i = (long)blockIdx.x*blockDim.x + threadIdx.x; i < total; i += (long)gridDim.x*blockDim.x) {
        const float v = mask[i];
        const int b = (int)(i >> 23);
        const unsigned bin = ordkey(v) >> 16;
        const unsigned thr = thrBin[b];
        if (bin > thr) {
            const unsigned pos = atomicAdd(&selCnt[b], 1u);
            if (pos < KSP) { selIdx[b*KSP + pos] = (unsigned)(i & (ET-1)); selVal[b*KSP + pos] = v; }
        } else if (bin == thr) {
            const unsigned pos = atomicAdd(&candCnt[b], 1u);
            if (pos < CAND_CAP) { candIdx[b*CAND_CAP + pos] = (unsigned)(i & (ET-1)); candVal[b*CAND_CAP + pos] = v; }
        }
    }
}

__global__ __launch_bounds__(256) void rank_select(
    const unsigned* __restrict__ candCnt, const unsigned* __restrict__ candIdx, const float* __restrict__ candVal,
    const unsigned* __restrict__ need, unsigned* __restrict__ selCnt,
    unsigned* __restrict__ selIdx, float* __restrict__ selVal)
{
    const int b = blockIdx.x;
    unsigned n = candCnt[b]; if (n > CAND_CAP) n = CAND_CAP;
    const unsigned k = need[b];
    for (unsigned i = threadIdx.x; i < n; i += blockDim.x) {
        const float vi = candVal[b*CAND_CAP + i];
        const unsigned xi = candIdx[b*CAND_CAP + i];
        unsigned rank = 0;
        for (unsigned j = 0; j < n; ++j) {
            const float vj = candVal[b*CAND_CAP + j];
            rank += (vj > vi) || ((vj == vi) && (candIdx[b*CAND_CAP + j] < xi));  // total order
        }
        if (rank < k) {
            const unsigned pos = atomicAdd(&selCnt[b], 1u);
            if (pos < KSP) { selIdx[b*KSP + pos] = xi; selVal[b*KSP + pos] = vi; }
        }
    }
}

// ---------------------------------------------------------------------------
// Finalize: zero output, compute up-dot only at selected points, scatter.
// ---------------------------------------------------------------------------
__global__ void zero_out(float4* __restrict__ p, int n) {
    const float4 z = make_float4(0.f, 0.f, 0.f, 0.f);
    for (int i = blockIdx.x*blockDim.x + threadIdx.x; i < n; i += gridDim.x*blockDim.x) p[i] = z;
}

__global__ __launch_bounds__(256) void up_scatter(
    const float* __restrict__ x, const float* __restrict__ Wup, const float* __restrict__ bup,
    const unsigned* __restrict__ selCnt, const unsigned* __restrict__ selIdx, const float* __restrict__ selVal,
    const float* __restrict__ bMax, const float* __restrict__ bZ, float* __restrict__ out)
{
    const int s = blockIdx.x;
    const int b = s >> 9;                 // KSP = 512
    const int j = s & (KSP - 1);
    unsigned cnt = selCnt[b]; if (cnt > KSP) cnt = KSP;
    if ((unsigned)j >= cnt) return;

    const unsigned idx = selIdx[b*KSP + j];
    const float    mv  = selVal[b*KSP + j];
    const int e = (int)(idx >> 11);       // T_ = 2^11
    const int t = (int)(idx & (T_ - 1));

    __shared__ float red[256];
    float part = 0.f;
    for (int c = threadIdx.x; c < C_; c += 256)
        part += Wup[(size_t)e*C_ + c] * x[(((size_t)b*C_ + c) << 11) + t];
    red[threadIdx.x] = part; __syncthreads();
    for (int ss = 128; ss > 0; ss >>= 1) { if (threadIdx.x < ss) red[threadIdx.x] += red[threadIdx.x + ss]; __syncthreads(); }
    if (threadIdx.x == 0) {
        const float up = red[0] + bup[e];
        const float sm = expf(mv - bMax[b]) / bZ[b];
        out[(((size_t)b*E_ + e) << 11) + t] = sm * up;
    }
}

// ---------------------------------------------------------------------------
extern "C" void kernel_launch(void* const* d_in, const int* in_sizes, int n_in,
                              void* d_out, int out_size, void* d_ws, size_t ws_size,
                              hipStream_t stream) {
    (void)in_sizes; (void)n_in; (void)out_size;
    const float* x    = (const float*)d_in[0];
    const float* Wup  = (const float*)d_in[1];
    const float* bup  = (const float*)d_in[2];
    const float* Wm   = (const float*)d_in[3];
    const float* bmsk = (const float*)d_in[4];
    float* out = (float*)d_out;

    char* ws = (char*)d_ws;
    size_t off = 0;
    auto take = [&](size_t bytes) -> char* {
        off = (off + 255) & ~(size_t)255;
        char* p = ws + off;
        off += bytes;
        return p;
    };
    unsigned* hist    = (unsigned*)take((size_t)B_ * 65536 * 4);   // 1 MB
    float*    wgMax   = (float*)   take((size_t)B_ * WGPB * 4);
    float*    wgSum   = (float*)   take((size_t)B_ * WGPB * 4);
    float*    bMax    = (float*)   take((size_t)B_ * 4);
    float*    bZ      = (float*)   take((size_t)B_ * 4);
    unsigned* ctrl    = (unsigned*)take(16 * 4);                   // selCnt|candCnt|thrBin|need
    unsigned* selIdx  = (unsigned*)take((size_t)B_ * KSP * 4);
    float*    selVal  = (float*)   take((size_t)B_ * KSP * 4);
    unsigned* candIdx = (unsigned*)take((size_t)B_ * CAND_CAP * 4);
    float*    candVal = (float*)   take((size_t)B_ * CAND_CAP * 4);
    // Pre-split operands (primary path): ~24 MB
    _Float16* WmH = (_Float16*)take((size_t)E_ * C_ * 2);
    _Float16* WmL = (_Float16*)take((size_t)E_ * C_ * 2);
    _Float16* xTH = (_Float16*)take((size_t)B_ * T_ * C_ * 2);
    _Float16* xTL = (_Float16*)take((size_t)B_ * T_ * C_ * 2);
    const bool usePre = (off <= ws_size);

    unsigned* selCnt  = ctrl;
    unsigned* candCnt = ctrl + 4;
    unsigned* thrBin  = ctrl + 8;
    unsigned* need    = ctrl + 12;

    zero_u32<<<256, 256, 0, stream>>>(hist, B_ * 65536);
    zero_u32<<<1, 64, 0, stream>>>(ctrl, 16);

    dim3 gg(T_/128, E_/128, B_);                                   // (16, 32, 4)
    if (usePre) {
        split_w<<<512, 256, 0, stream>>>(Wm, WmH, WmL, E_ * C_);
        split_transpose_x<<<dim3(T_/32, C_/32, B_), 256, 0, stream>>>(x, xTH, xTL);
        mask_gemm_async<<<gg, 256, 0, stream>>>(WmH, WmL, xTH, xTL, bmsk, out, wgMax, wgSum);
    } else {
        mask_gemm<<<gg, 256, 0, stream>>>(x, Wm, bmsk, out, wgMax, wgSum);
    }
    softmax_stats<<<B_, 256, 0, stream>>>(wgMax, wgSum, bMax, bZ);

    hist_kernel<<<2048, 256, 0, stream>>>(out, hist);
    find_threshold<<<B_, 256, 0, stream>>>(hist, thrBin, need);
    collect<<<2048, 256, 0, stream>>>(out, thrBin, selCnt, selIdx, selVal, candCnt, candIdx, candVal);
    rank_select<<<B_, 256, 0, stream>>>(candCnt, candIdx, candVal, need, selCnt, selIdx, selVal);

    zero_out<<<4096, 256, 0, stream>>>((float4*)out, B_ * ET / 4);
    up_scatter<<<B_ * KSP, 256, 0, stream>>>(x, Wup, bup, selCnt, selIdx, selVal, bMax, bZ, out);
}
